// MultiHeadAttention_5832565588652
// MI455X (gfx1250) — matrix operand order repaired and checked by reading, not compiled, not run on hardware
//
#include <hip/hip_runtime.h>
#include <stdint.h>

// ---------------------------------------------------------------------------
// MHA: x@w_qkv -> flash attention -> @w_out + b_out
// B=8 S=1024 D=768 H=12 DQK=DV=64 DHEAD=192 NQKV=2304 DOUT=768, M=B*S=8192
// All matmuls via v_wmma_f32_16x16x32_bf16 (fp32 accumulate).
// Round 4: preload all fragments before WMMA burst (avoid per-WMMA dscnt-0
//          stalls; let WMMA overlap with in-flight ds_load_b128).
// ---------------------------------------------------------------------------

typedef __attribute__((ext_vector_type(16))) __bf16 v16bf;
typedef __attribute__((ext_vector_type(8)))  float  v8f;
typedef __attribute__((ext_vector_type(4)))  int    v4i;

#define WMMA_BF16(a, b, c) \
  __builtin_amdgcn_wmma_f32_16x16x32_bf16(false, (a), false, (b), (short)0, (c), false, false)

struct U256 { uint4 lo, hi; };

__device__ __forceinline__ v16bf make_frag(uint4 lo, uint4 hi) {
  U256 t; t.lo = lo; t.hi = hi;
  return __builtin_bit_cast(v16bf, t);
}

// native f32 -> bf16 (v_cvt_pk_bf16_f32), RTNE
__device__ __forceinline__ unsigned short f2bf(float f) {
  return __builtin_bit_cast(unsigned short, (__bf16)f);
}

// ---- async global->LDS copy (ASYNCcnt path), with synchronous fallback ----
#if defined(__has_builtin)
#if __has_builtin(__builtin_amdgcn_global_load_async_to_lds_b128)
#define HAVE_ASYNC_LDS 1
#endif
#endif

typedef __attribute__((address_space(1))) v4i* gv4i_p;  // global b128 ptr
typedef __attribute__((address_space(3))) v4i* lv4i_p;  // LDS b128 ptr

__device__ __forceinline__ void async_copy_b128(const unsigned short* g, unsigned short* l) {
#ifdef HAVE_ASYNC_LDS
  __builtin_amdgcn_global_load_async_to_lds_b128((gv4i_p)g, (lv4i_p)l, 0, 0);
#else
  *(uint4*)l = *(const uint4*)g;
#endif
}

__device__ __forceinline__ void async_wait_all() {
#ifdef HAVE_ASYNC_LDS
#if __has_builtin(__builtin_amdgcn_s_wait_asynccnt)
  __builtin_amdgcn_s_wait_asynccnt(0);
#else
  asm volatile("s_wait_asynccnt 0" ::: "memory");
#endif
#endif
}

// ---------------------------------------------------------------------------
// Kernel 1: qkv = x @ w_qkv  (fp32 in, bf16 out)  M=8192 K=768 N=2304
// Block tile 128x128; 8 waves as 4(M)x2(N); wave tile 32x64 -> 8 WMMA/K-step.
// ---------------------------------------------------------------------------
__global__ __launch_bounds__(256) void qkv_gemm_kernel(
    const float* __restrict__ x, const float* __restrict__ w,
    unsigned short* __restrict__ out /* bf16 [8192][2304] */) {
  const int N = 2304, K = 768;
  __shared__ unsigned short lds_a[128 * 32];  // [row][k]
  __shared__ unsigned short lds_b[128 * 32];  // [n][k]

  const int n0 = blockIdx.x * 128;
  const int m0 = blockIdx.y * 128;
  const int t = threadIdx.x;
  const int lane = t & 31, wid = t >> 5;
  const int l16 = lane & 15, half = lane >> 4;
  const int wm = (wid & 3) * 32, wn = (wid >> 2) * 64;

  v8f acc[2][4] = {};

  for (int k0 = 0; k0 < K; k0 += 32) {
    // stage A 128x32 fp32->bf16 (packed b64 stores)
    for (int j = 0; j < 4; j++) {
      int e = t * 16 + j * 4;
      int row = e >> 5, col = e & 31;
      float4 f = *(const float4*)&x[(size_t)(m0 + row) * K + k0 + col];
      ushort4 h;
      h.x = f2bf(f.x); h.y = f2bf(f.y); h.z = f2bf(f.z); h.w = f2bf(f.w);
      *(ushort4*)&lds_a[row * 32 + col] = h;
    }
    // stage B 32k x 128n transposed to [n][k]
    {
      int n = t & 127, kk = t >> 7;  // kk in 0..1
      for (int i = 0; i < 16; i++) {
        int k = kk * 16 + i;
        lds_b[n * 32 + k] = f2bf(w[(size_t)(k0 + k) * N + n0 + n]);
      }
    }
    __syncthreads();

    // preload ALL fragments, then issue the 8-WMMA burst
    v16bf a0 = make_frag(*(const uint4*)&lds_a[(wm + l16) * 32 + 8 * half],
                         *(const uint4*)&lds_a[(wm + l16) * 32 + 16 + 8 * half]);
    v16bf a1 = make_frag(*(const uint4*)&lds_a[(wm + 16 + l16) * 32 + 8 * half],
                         *(const uint4*)&lds_a[(wm + 16 + l16) * 32 + 16 + 8 * half]);
    v16bf bf[4];
    for (int c = 0; c < 4; c++) {
      int bn = wn + c * 16 + l16;
      bf[c] = make_frag(*(const uint4*)&lds_b[bn * 32 + 8 * half],
                        *(const uint4*)&lds_b[bn * 32 + 16 + 8 * half]);
    }
    for (int c = 0; c < 4; c++) {
      acc[0][c] = WMMA_BF16(a0, bf[c], acc[0][c]);
      acc[1][c] = WMMA_BF16(a1, bf[c], acc[1][c]);
    }
    __syncthreads();
  }

  for (int i = 0; i < 2; i++)
    for (int c = 0; c < 4; c++)
      for (int r = 0; r < 8; r++) {
        int grow = m0 + wm + i * 16 + r + 8 * half;
        int gcol = n0 + wn + c * 16 + l16;
        out[(size_t)grow * N + gcol] = f2bf(acc[i][c][r]);
      }
}

// ---------------------------------------------------------------------------
// Kernel 2: flash attention per (b, h, 128-query block). 8 waves x 16 rows.
// ---------------------------------------------------------------------------
__global__ __launch_bounds__(256) void attn_kernel(
    const unsigned short* __restrict__ qkv,  // bf16 [8192][2304]
    unsigned short* __restrict__ attn) {     // bf16 [8192][768]
  __shared__ unsigned short q_lds[128 * 64];   // [qrow][dqk]
  __shared__ unsigned short k_lds[32 * 64];    // [key][dqk]
  __shared__ unsigned short v_lds[64 * 32];    // [vdim][key]
  __shared__ unsigned short p_lds[8][16 * 32]; // per-wave [row][key]

  const int t = threadIdx.x;
  const int lane = t & 31, wid = t >> 5;
  const int l16 = lane & 15, half = lane >> 4;

  const int qb = blockIdx.x & 7;
  const int h  = (blockIdx.x >> 3) % 12;
  const int b  = blockIdx.x / 96;

  const size_t rs = 2304;
  const size_t base = (size_t)b * 1024 * rs + (size_t)h * 192;

  // stage Q block via async copies: 128 rows x 64 bf16
  {
    int row = t >> 1, part = t & 1;
    for (int j = 0; j < 4; j++) {
      int off = part * 32 + j * 8;
      async_copy_b128(&qkv[base + (size_t)(qb * 128 + row) * rs + off],
                      &q_lds[row * 64 + off]);
    }
  }
  async_wait_all();
  __syncthreads();

  const int qrow = wid * 16 + l16;
  v16bf qa[2];
  for (int kc = 0; kc < 2; kc++)
    qa[kc] = make_frag(*(const uint4*)&q_lds[qrow * 64 + kc * 32 + 8 * half],
                       *(const uint4*)&q_lds[qrow * 64 + kc * 32 + 16 + 8 * half]);

  float m_i[8], l_i[8];
  v8f o[4] = {};
  for (int r = 0; r < 8; r++) { m_i[r] = -3.0e38f; l_i[r] = 0.f; }
  const float scale = 0.125f;  // 64^-0.5

  for (int jb = 0; jb < 32; jb++) {
    // K tile (pure copy) via async path
    {
      int key = t >> 3, part = t & 7;
      async_copy_b128(&qkv[base + (size_t)(jb * 32 + key) * rs + 64 + part * 8],
                      &k_lds[key * 64 + part * 8]);
    }
    // V tile transposed to [vdim][key] (manual, needs transpose)
    {
      int key = t & 31, d0 = t >> 5;
      for (int i = 0; i < 8; i++) {
        int dim = d0 + i * 8;
        v_lds[dim * 32 + key] = qkv[base + (size_t)(jb * 32 + key) * rs + 128 + dim];
      }
    }
    async_wait_all();
    __syncthreads();

    // S = Q @ K^T : preload all 4 K fragments, then 4 WMMAs
    v16bf kb[2][2];
    for (int c = 0; c < 2; c++) {
      int kn = c * 16 + l16;
      for (int kc = 0; kc < 2; kc++)
        kb[c][kc] = make_frag(*(const uint4*)&k_lds[kn * 64 + kc * 32 + 8 * half],
                              *(const uint4*)&k_lds[kn * 64 + kc * 32 + 16 + 8 * half]);
    }
    v8f s[2] = {};
    for (int c = 0; c < 2; c++)
      for (int kc = 0; kc < 2; kc++)
        s[c] = WMMA_BF16(qa[kc], kb[c][kc], s[c]);

    // online softmax (rows live in 16-lane halves of the wave)
    float f[8];
    for (int r = 0; r < 8; r++) {
      float v0 = s[0][r] * scale, v1 = s[1][r] * scale;
      float rm = fmaxf(v0, v1);
      rm = fmaxf(rm, __shfl_xor(rm, 1, 32));
      rm = fmaxf(rm, __shfl_xor(rm, 2, 32));
      rm = fmaxf(rm, __shfl_xor(rm, 4, 32));
      rm = fmaxf(rm, __shfl_xor(rm, 8, 32));
      float mn = fmaxf(m_i[r], rm);
      f[r] = __expf(m_i[r] - mn);
      m_i[r] = mn;
      float p0 = __expf(v0 - mn), p1 = __expf(v1 - mn);
      s[0][r] = p0; s[1][r] = p1;
      float sum = p0 + p1;
      sum += __shfl_xor(sum, 1, 32);
      sum += __shfl_xor(sum, 2, 32);
      sum += __shfl_xor(sum, 4, 32);
      sum += __shfl_xor(sum, 8, 32);
      l_i[r] = l_i[r] * f[r] + sum;
    }
    for (int n = 0; n < 4; n++)
      for (int r = 0; r < 8; r++)
        o[n][r] = o[n][r] * f[r];

    // re-layout P (C layout -> A fragment) through per-wave LDS
    unsigned short* pw = p_lds[wid];
    for (int c = 0; c < 2; c++)
      for (int r = 0; r < 8; r++)
        pw[(r + 8 * half) * 32 + c * 16 + l16] = f2bf(s[c][r]);
    asm volatile("s_wait_dscnt 0" ::: "memory");   // wave-local LDS RAW

    // O += P @ V : preload P + all 4 V fragments, then 4 WMMAs
    v16bf pa = make_frag(*(const uint4*)&pw[l16 * 32 + 8 * half],
                         *(const uint4*)&pw[l16 * 32 + 16 + 8 * half]);
    v16bf vb[4];
    for (int n = 0; n < 4; n++) {
      int vn = n * 16 + l16;
      vb[n] = make_frag(*(const uint4*)&v_lds[vn * 32 + 8 * half],
                        *(const uint4*)&v_lds[vn * 32 + 16 + 8 * half]);
    }
    for (int n = 0; n < 4; n++)
      o[n] = WMMA_BF16(pa, vb[n], o[n]);
    __syncthreads();
  }

  // normalize and store bf16 [token][h*64 + vdim]
  for (int r = 0; r < 8; r++) {
    float inv = 1.0f / l_i[r];
    int srow = b * 1024 + qb * 128 + wid * 16 + r + 8 * half;
    for (int n = 0; n < 4; n++)
      attn[(size_t)srow * 768 + h * 64 + n * 16 + l16] = f2bf(o[n][r] * inv);
  }
}

// ---------------------------------------------------------------------------
// Kernel 3: out = attn @ w_out + b_out  (bf16 A, fp32 W/out) M=8192 K=N=768
// Block tile 128x128; wave tile 32x64; A staged via async copies.
// ---------------------------------------------------------------------------
__global__ __launch_bounds__(256) void out_gemm_kernel(
    const unsigned short* __restrict__ a /* bf16 [8192][768] */,
    const float* __restrict__ w, const float* __restrict__ bias,
    float* __restrict__ out) {
  const int N = 768, K = 768;
  __shared__ unsigned short lds_a[128 * 32];
  __shared__ unsigned short lds_b[128 * 32];

  const int n0 = blockIdx.x * 128;
  const int m0 = blockIdx.y * 128;
  const int t = threadIdx.x;
  const int lane = t & 31, wid = t >> 5;
  const int l16 = lane & 15, half = lane >> 4;
  const int wm = (wid & 3) * 32, wn = (wid >> 2) * 64;

  v8f acc[2][4] = {};

  for (int k0 = 0; k0 < K; k0 += 32) {
    // A tile already bf16: async copy 128x32 (2 x b128 per thread)
    for (int j = 0; j < 2; j++) {
      int e = t * 16 + j * 8;
      int row = e >> 5, col = e & 31;
      async_copy_b128(&a[(size_t)(m0 + row) * K + k0 + col], &lds_a[row * 32 + col]);
    }
    // B tile 32k x 128n, fp32 -> bf16, transposed to [n][k]
    {
      int n = t & 127, kk = t >> 7;
      for (int i = 0; i < 16; i++) {
        int k = kk * 16 + i;
        lds_b[n * 32 + k] = f2bf(w[(size_t)(k0 + k) * N + n0 + n]);
      }
    }
    async_wait_all();
    __syncthreads();

    v16bf a0 = make_frag(*(const uint4*)&lds_a[(wm + l16) * 32 + 8 * half],
                         *(const uint4*)&lds_a[(wm + l16) * 32 + 16 + 8 * half]);
    v16bf a1 = make_frag(*(const uint4*)&lds_a[(wm + 16 + l16) * 32 + 8 * half],
                         *(const uint4*)&lds_a[(wm + 16 + l16) * 32 + 16 + 8 * half]);
    v16bf bf[4];
    for (int c = 0; c < 4; c++) {
      int bn = wn + c * 16 + l16;
      bf[c] = make_frag(*(const uint4*)&lds_b[bn * 32 + 8 * half],
                        *(const uint4*)&lds_b[bn * 32 + 16 + 8 * half]);
    }
    for (int c = 0; c < 4; c++) {
      acc[0][c] = WMMA_BF16(a0, bf[c], acc[0][c]);
      acc[1][c] = WMMA_BF16(a1, bf[c], acc[1][c]);
    }
    __syncthreads();
  }

  for (int i = 0; i < 2; i++)
    for (int c = 0; c < 4; c++)
      for (int r = 0; r < 8; r++) {
        int grow = m0 + wm + i * 16 + r + 8 * half;
        int gcol = n0 + wn + c * 16 + l16;
        out[(size_t)grow * N + gcol] = acc[i][c][r] + bias[gcol];
      }
}

// ---------------------------------------------------------------------------
extern "C" void kernel_launch(void* const* d_in, const int* in_sizes, int n_in,
                              void* d_out, int out_size, void* d_ws, size_t ws_size,
                              hipStream_t stream) {
  const float* x     = (const float*)d_in[0];
  const float* w_qkv = (const float*)d_in[1];
  const float* w_out = (const float*)d_in[2];
  const float* b_out = (const float*)d_in[3];
  float* out = (float*)d_out;

  unsigned short* qkv_ws  = (unsigned short*)d_ws;         // 8192*2304 bf16
  unsigned short* attn_ws = qkv_ws + (size_t)8192 * 2304;  // 8192*768 bf16

  qkv_gemm_kernel<<<dim3(2304 / 128, 8192 / 128), 256, 0, stream>>>(x, w_qkv, qkv_ws);
  attn_kernel<<<dim3(8 * 12 * 8), 256, 0, stream>>>(qkv_ws, attn_ws);
  out_gemm_kernel<<<dim3(768 / 128, 8192 / 128), 256, 0, stream>>>(attn_ws, w_out, b_out, out);
}
